// MultiheadDiffAttn_62414464745484
// MI455X (gfx1250) — matrix operand order repaired and checked by reading, not compiled, not run on hardware
//
#include <hip/hip_runtime.h>
#include <hip/hip_bf16.h>

// ---------------------------------------------------------------------------
// MultiheadDiffAttn for MI455X (gfx1250, wave32, WMMA bf16 16x16x32)
//   B=2, T=2048, D=1024, H=16, KV=8 (GQA rep=2), DH=64, non-causal softmax
// ---------------------------------------------------------------------------

#define B_  2
#define T_  2048
#define D_  1024
#define H_  16
#define KV_ 8
#define DH_ 64

#define LAMBDA_INIT 0.7836057665316245f
#define OUT_SCALE   (1.0f - LAMBDA_INIT)

typedef __bf16 bf16_t;
typedef __attribute__((ext_vector_type(16))) __bf16 v16bf;
typedef __attribute__((ext_vector_type(8)))  __bf16 v8bf;
typedef __attribute__((ext_vector_type(8)))  float  v8f;
typedef __attribute__((ext_vector_type(4)))  float  v4f;

struct alignas(16) U4 { unsigned int x, y, z, w; };
union AF { U4 u[2]; v16bf v; };

// D = A(16x32) x B(32x16) + C  with f32 accumulate
__device__ __forceinline__ v8f wmma_bf16(v16bf a, v16bf b, v8f c) {
  return __builtin_amdgcn_wmma_f32_16x16x32_bf16(
      /*neg_a=*/false, a, /*neg_b=*/false, b,
      /*c_mod=*/(short)0, c, /*reuse_a=*/false, /*reuse_b=*/false);
}

// A-fragment (16x32, row-major source, leading dim `ld` elements):
//  lane<16: row=lane,    K = 0..7  (v0-3) and 16..23 (v4-7)
//  lane>=16: row=lane-16, K = 8..15 (v0-3) and 24..31 (v4-7)
__device__ __forceinline__ v16bf load_a_frag(const bf16_t* p0, int ld) {
  const int lane = threadIdx.x & 31;
  const bf16_t* p = p0 + (size_t)(lane & 15) * ld + ((lane >> 4) << 3);
  AF t;
  t.u[0] = *(const U4*)(p);
  t.u[1] = *(const U4*)(p + 16);
  return t.v;
}

// B-fragment (32x16) built from a ROW-MAJOR [N x K] source (i.e. B = src^T):
//  lane<16: col=lane, K = 0..7 and 16..23; lane>=16: col=lane-16, K = 8..15 and 24..31
__device__ __forceinline__ v16bf load_b_frag(const bf16_t* p0, int ld) {
  const int lane = threadIdx.x & 31;
  const bf16_t* p = p0 + (size_t)(lane & 15) * ld + ((lane >> 4) << 3);
  AF t;
  t.u[0] = *(const U4*)(p);
  t.u[1] = *(const U4*)(p + 16);
  return t.v;
}

// ---------------------------------------------------------------------------
// 16-lane butterfly reductions done fully in the VALU with DPP16
// (no ds_bpermute, no s_wait_dscnt stalls in the softmax hot loop).
//   xor1: quad_perm [1,0,3,2] = 0xB1
//   xor2: quad_perm [2,3,0,1] = 0x4E
//   xor4: row_half_mirror     = 0x141
//   xor8: row_mirror          = 0x140
// DPP ctrl must be an integer-constant at the builtin call site -> template.
// ---------------------------------------------------------------------------
#define DPP_QUAD_XOR1    0xB1
#define DPP_QUAD_XOR2    0x4E
#define DPP_ROW_HMIRROR  0x141
#define DPP_ROW_MIRROR   0x140

template <int CTRL>
__device__ __forceinline__ float dpp_movf(float v) {
  return __int_as_float(__builtin_amdgcn_update_dpp(
      0, __float_as_int(v), CTRL, 0xf, 0xf, true));
}
__device__ __forceinline__ float hmax16(float v) {
  v = fmaxf(v, dpp_movf<DPP_QUAD_XOR1>(v));
  v = fmaxf(v, dpp_movf<DPP_QUAD_XOR2>(v));
  v = fmaxf(v, dpp_movf<DPP_ROW_HMIRROR>(v));
  v = fmaxf(v, dpp_movf<DPP_ROW_MIRROR>(v));
  return v;
}
__device__ __forceinline__ float hsum16(float v) {
  v += dpp_movf<DPP_QUAD_XOR1>(v);
  v += dpp_movf<DPP_QUAD_XOR2>(v);
  v += dpp_movf<DPP_ROW_HMIRROR>(v);
  v += dpp_movf<DPP_ROW_MIRROR>(v);
  return v;
}

__device__ __forceinline__ void cvt_store(float* p, float v)  { *p = v; }
__device__ __forceinline__ void cvt_store(bf16_t* p, float v) { *p = (bf16_t)v; }

// ---------------------------------------------------------------------------
// Vectorized f32 -> bf16: 8 elements/thread (2x b128 load, 1x b128 store)
// ---------------------------------------------------------------------------
__global__ void f32_to_bf16_kernel(const float* __restrict__ x,
                                   bf16_t* __restrict__ y, int n) {
  int i = (blockIdx.x * blockDim.x + threadIdx.x) * 8;
  if (i >= n) return;
  v4f a = *(const v4f*)(x + i);
  v4f b = *(const v4f*)(x + i + 4);
  v8bf o;
#pragma unroll
  for (int j = 0; j < 4; ++j) {
    o[j]     = (bf16_t)a[j];
    o[4 + j] = (bf16_t)b[j];
  }
  *(v8bf*)(y + i) = o;
}

// ---------------------------------------------------------------------------
// GEMM: C[M,N] = A[M,K] * Bt[N,K]^T   (torch Linear semantics: x @ W.T)
// One wave computes a 64x64 C tile via a 4x4 grid of 16x16x32 WMMAs.
// ---------------------------------------------------------------------------
template <typename OutT>
__global__ __launch_bounds__(256) void gemm_bf16_wmma(
    const bf16_t* __restrict__ A, const bf16_t* __restrict__ Bt,
    OutT* __restrict__ C, int M, int N, int K) {
  const int wave   = blockIdx.x * 8 + (threadIdx.x >> 5);
  const int ntiles = N >> 6;
  const int m0 = (wave / ntiles) << 6;
  const int n0 = (wave % ntiles) << 6;
  if (m0 >= M) return;

  v8f zero = {};
  v8f acc[4][4];
#pragma unroll
  for (int i = 0; i < 4; ++i)
#pragma unroll
    for (int j = 0; j < 4; ++j) acc[i][j] = zero;

  for (int k0 = 0; k0 < K; k0 += 32) {
    v16bf a[4], b[4];
#pragma unroll
    for (int i = 0; i < 4; ++i)
      a[i] = load_a_frag(A + (size_t)(m0 + 16 * i) * K + k0, K);
#pragma unroll
    for (int j = 0; j < 4; ++j)
      b[j] = load_b_frag(Bt + (size_t)(n0 + 16 * j) * K + k0, K);
#pragma unroll
    for (int i = 0; i < 4; ++i)
#pragma unroll
      for (int j = 0; j < 4; ++j)
        acc[i][j] = wmma_bf16(a[i], b[j], acc[i][j]);
  }

  const int lane = threadIdx.x & 31;
  const int col  = lane & 15;
  const int rb   = (lane >> 4) << 3;
#pragma unroll
  for (int i = 0; i < 4; ++i)
#pragma unroll
    for (int j = 0; j < 4; ++j)
#pragma unroll
      for (int r = 0; r < 8; ++r) {
        size_t idx = (size_t)(m0 + 16 * i + rb + r) * N + (n0 + 16 * j + col);
        cvt_store(C + idx, acc[i][j][r]);
      }
}

// ---------------------------------------------------------------------------
// Fused differential flash-attention.
// qp: [B, T, 2*H*DH]  (q1 cols [0,1024), q2 cols [1024,2048), head h at h*64)
// kp: [B, T, 2*KV*DH] (k1 cols [0,512),  k2 cols [512,1024))
// vp: [B, T, KV*DH]
// One wave: 16 query rows of one (b,h); streams keys in chunks of 32 with
// two online softmaxes; out = acc1/l1 - lam*acc2/l2 -> attn f32 [B,T,D].
// ---------------------------------------------------------------------------
__global__ __launch_bounds__(256) void diff_attn_kernel(
    const bf16_t* __restrict__ qp, const bf16_t* __restrict__ kp,
    const bf16_t* __restrict__ vp, const float* __restrict__ lam_ptr,
    float* __restrict__ attn) {
  __shared__ alignas(16) bf16_t lds_vT[8][64 * 32];  // V^T chunk per wave
  __shared__ alignas(16) bf16_t lds_p1[8][16 * 32];  // prob tiles (A-layout src)
  __shared__ alignas(16) bf16_t lds_p2[8][16 * 32];

  const int w    = threadIdx.x >> 5;
  const int lane = threadIdx.x & 31;
  const int wave = blockIdx.x * 8 + w;
  const int TQ   = T_ / 16;
  const int b    = wave / (H_ * TQ);
  const int rem  = wave - b * (H_ * TQ);
  const int h    = rem / TQ;
  const int t0   = (rem - h * TQ) * 16;
  const int kv   = h >> 1;  // repeat_kv interleave: q head h -> kv head h/2
  const float lam = *lam_ptr;

  const int LDQ = 2 * H_ * DH_;   // 2048
  const int LDK = 2 * KV_ * DH_;  // 1024
  const int LDV = KV_ * DH_;      // 512

  // Query fragments (persist across the key loop): 2 K-steps each for q1,q2
  const bf16_t* qbase = qp + ((size_t)b * T_ + t0) * LDQ;
  v16bf q1f[2], q2f[2];
#pragma unroll
  for (int kk = 0; kk < 2; ++kk) {
    q1f[kk] = load_a_frag(qbase + h * DH_ + kk * 32, LDQ);
    q2f[kk] = load_a_frag(qbase + H_ * DH_ + h * DH_ + kk * 32, LDQ);
  }

  const bf16_t* kbase = kp + (size_t)b * T_ * LDK;
  const bf16_t* vbase = vp + (size_t)b * T_ * LDV + kv * DH_;

  v8f zero = {};
  v8f acc1[4], acc2[4];
  float m1[8], l1[8], m2[8], l2[8];
#pragma unroll
  for (int n = 0; n < 4; ++n) { acc1[n] = zero; acc2[n] = zero; }
#pragma unroll
  for (int j = 0; j < 8; ++j) {
    m1[j] = -3.0e38f; m2[j] = -3.0e38f; l1[j] = 0.f; l2[j] = 0.f;
  }

  const int col = lane & 15;
  const int rb  = (lane >> 4) << 3;

  for (int s0 = 0; s0 < T_; s0 += 32) {
    // ---- stage V chunk transposed into LDS: lds_vT[d][k] = v[s0+k][d] ----
    {
      const bf16_t* vrow = vbase + (size_t)(s0 + lane) * LDV;
#pragma unroll
      for (int d = 0; d < 64; d += 8) {
        union { U4 u; bf16_t hh[8]; } t;
        t.u = *(const U4*)(vrow + d);
#pragma unroll
        for (int dd = 0; dd < 8; ++dd)
          lds_vT[w][(d + dd) * 32 + lane] = t.hh[dd];
      }
    }

    // ---- scores: s1/s2 are [16 rows x 32 keys] as two 16x16 C tiles ----
    v8f s1[2], s2[2];
    s1[0] = zero; s1[1] = zero; s2[0] = zero; s2[1] = zero;
#pragma unroll
    for (int c = 0; c < 2; ++c) {
#pragma unroll
      for (int kk = 0; kk < 2; ++kk) {
        v16bf bk1 = load_b_frag(kbase + (size_t)(s0 + 16 * c) * LDK +
                                    kv * DH_ + kk * 32, LDK);
        s1[c] = wmma_bf16(q1f[kk], bk1, s1[c]);
        v16bf bk2 = load_b_frag(kbase + (size_t)(s0 + 16 * c) * LDK +
                                    KV_ * DH_ + kv * DH_ + kk * 32, LDK);
        s2[c] = wmma_bf16(q2f[kk], bk2, s2[c]);
      }
    }

    // ---- online softmax update (row-wise; rows live in 16-lane halves;
    //      reductions are pure-VALU DPP butterflies) ----
    float p1v[2][8], p2v[2][8];
#pragma unroll
    for (int j = 0; j < 8; ++j) {
      // softmax 1
      float rm = hmax16(fmaxf(s1[0][j], s1[1][j]));
      float mn = fmaxf(m1[j], rm);
      float sc = __expf(m1[j] - mn);
      float e0 = __expf(s1[0][j] - mn);
      float e1 = __expf(s1[1][j] - mn);
      l1[j] = l1[j] * sc + hsum16(e0 + e1);
      m1[j] = mn;
#pragma unroll
      for (int n = 0; n < 4; ++n) acc1[n][j] *= sc;
      p1v[0][j] = e0; p1v[1][j] = e1;
      // softmax 2
      rm = hmax16(fmaxf(s2[0][j], s2[1][j]));
      mn = fmaxf(m2[j], rm);
      sc = __expf(m2[j] - mn);
      e0 = __expf(s2[0][j] - mn);
      e1 = __expf(s2[1][j] - mn);
      l2[j] = l2[j] * sc + hsum16(e0 + e1);
      m2[j] = mn;
#pragma unroll
      for (int n = 0; n < 4; ++n) acc2[n][j] *= sc;
      p2v[0][j] = e0; p2v[1][j] = e1;
    }

    // ---- write prob tiles to LDS row-major [16 x 32] (same-wave, DS in-order)
#pragma unroll
    for (int c = 0; c < 2; ++c)
#pragma unroll
      for (int j = 0; j < 8; ++j) {
        lds_p1[w][(rb + j) * 32 + 16 * c + col] = (bf16_t)p1v[c][j];
        lds_p2[w][(rb + j) * 32 + 16 * c + col] = (bf16_t)p2v[c][j];
      }

    // ---- PV: acc[16x64] += P[16x32] @ V[32x64] ----
    v16bf pa1 = load_a_frag(&lds_p1[w][0], 32);
    v16bf pa2 = load_a_frag(&lds_p2[w][0], 32);
    v16bf bv[4];
#pragma unroll
    for (int n = 0; n < 4; ++n)
      bv[n] = load_b_frag(&lds_vT[w][(16 * n) * 32], 32);
#pragma unroll
    for (int n = 0; n < 4; ++n) acc1[n] = wmma_bf16(pa1, bv[n], acc1[n]);
#pragma unroll
    for (int n = 0; n < 4; ++n) acc2[n] = wmma_bf16(pa2, bv[n], acc2[n]);
  }

  // ---- finalize: out = acc1/l1 - lam*acc2/l2, scatter into [B,T,D] ----
#pragma unroll
  for (int n = 0; n < 4; ++n)
#pragma unroll
    for (int j = 0; j < 8; ++j) {
      int t = t0 + rb + j;
      int d = h * DH_ + 16 * n + col;
      float o = acc1[n][j] / l1[j] - lam * (acc2[n][j] / l2[j]);
      attn[((size_t)b * T_ + t) * D_ + d] = o;
    }
}

// ---------------------------------------------------------------------------
// RMSNorm * g * (1 - lambda_init), f32 -> bf16. One wave per 1024-elem row.
// ---------------------------------------------------------------------------
__global__ __launch_bounds__(256) void rmsnorm_scale_kernel(
    const float* __restrict__ x, const float* __restrict__ g,
    bf16_t* __restrict__ y) {
  const int row  = blockIdx.x * 8 + (threadIdx.x >> 5);
  const int lane = threadIdx.x & 31;
  const float* xr = x + (size_t)row * D_;
  float ss = 0.f;
#pragma unroll
  for (int i = lane; i < D_; i += 32) { float v = xr[i]; ss = fmaf(v, v, ss); }
  // full-wave reduction: 16-lane DPP butterfly + one cross-half shuffle
  ss += dpp_movf<DPP_QUAD_XOR1>(ss);
  ss += dpp_movf<DPP_QUAD_XOR2>(ss);
  ss += dpp_movf<DPP_ROW_HMIRROR>(ss);
  ss += dpp_movf<DPP_ROW_MIRROR>(ss);
  ss += __shfl_xor(ss, 16, 32);  // cross-half (single ds op per row)
  const float sc = rsqrtf(ss * (1.0f / D_) + 1e-5f) * OUT_SCALE;
  bf16_t* yr = y + (size_t)row * D_;
#pragma unroll
  for (int i = lane; i < D_; i += 32) yr[i] = (bf16_t)(xr[i] * sc * g[i]);
}

// ---------------------------------------------------------------------------
// Workspace layout (bytes)
// ---------------------------------------------------------------------------
static const size_t OFF_QBF    = 0;                         //  8 MB bf16 q
static const size_t OFF_KBF    = OFF_QBF  + 8388608;        //  8 MB bf16 k
static const size_t OFF_VBF    = OFF_KBF  + 8388608;        //  8 MB bf16 v
static const size_t OFF_WQBF   = OFF_VBF  + 8388608;        //  4 MB
static const size_t OFF_WKBF   = OFF_WQBF + 4194304;        //  2 MB
static const size_t OFF_WVBF   = OFF_WKBF + 2097152;        //  1 MB
static const size_t OFF_WOBF   = OFF_WVBF + 1048576;        //  2 MB
static const size_t OFF_QP     = OFF_WOBF + 2097152;        // 16 MB [B,T,2048]
static const size_t OFF_KP     = OFF_QP   + 16777216;       //  8 MB [B,T,1024]
static const size_t OFF_VP     = OFF_KP   + 8388608;        //  4 MB [B,T,512]
static const size_t OFF_ATTNF  = OFF_VP   + 4194304;        // 16 MB f32 [B,T,D]
static const size_t OFF_ATTNBF = OFF_ATTNF+ 16777216;       //  8 MB bf16

extern "C" void kernel_launch(void* const* d_in, const int* in_sizes, int n_in,
                              void* d_out, int out_size, void* d_ws, size_t ws_size,
                              hipStream_t stream) {
  (void)in_sizes; (void)n_in; (void)out_size; (void)ws_size;
  const float* q   = (const float*)d_in[0];
  const float* k   = (const float*)d_in[1];
  const float* v   = (const float*)d_in[2];
  const float* Wq  = (const float*)d_in[3];
  const float* Wk  = (const float*)d_in[4];
  const float* Wv  = (const float*)d_in[5];
  const float* Wo  = (const float*)d_in[6];
  const float* lam = (const float*)d_in[7];
  const float* g   = (const float*)d_in[8];
  float* out = (float*)d_out;

  char* ws = (char*)d_ws;
  bf16_t* qbf    = (bf16_t*)(ws + OFF_QBF);
  bf16_t* kbf    = (bf16_t*)(ws + OFF_KBF);
  bf16_t* vbf    = (bf16_t*)(ws + OFF_VBF);
  bf16_t* Wqbf   = (bf16_t*)(ws + OFF_WQBF);
  bf16_t* Wkbf   = (bf16_t*)(ws + OFF_WKBF);
  bf16_t* Wvbf   = (bf16_t*)(ws + OFF_WVBF);
  bf16_t* Wobf   = (bf16_t*)(ws + OFF_WOBF);
  bf16_t* qpbf   = (bf16_t*)(ws + OFF_QP);
  bf16_t* kpbf   = (bf16_t*)(ws + OFF_KP);
  bf16_t* vpbf   = (bf16_t*)(ws + OFF_VP);
  float*  attnf  = (float*)(ws + OFF_ATTNF);
  bf16_t* attnbf = (bf16_t*)(ws + OFF_ATTNBF);

  const int M = B_ * T_;  // 4096

  // 1) f32 -> bf16 converts (8 elems/thread, b128 traffic)
  auto cvt = [&](const float* src, bf16_t* dst, int n) {
    int threads = n / 8;
    f32_to_bf16_kernel<<<(threads + 255) / 256, 256, 0, stream>>>(src, dst, n);
  };
  cvt(q,  qbf,  M * D_);
  cvt(k,  kbf,  M * D_);
  cvt(v,  vbf,  M * D_);
  cvt(Wq, Wqbf, 2 * H_ * DH_ * D_);
  cvt(Wk, Wkbf, 2 * KV_ * DH_ * D_);
  cvt(Wv, Wvbf, KV_ * DH_ * D_);
  cvt(Wo, Wobf, D_ * D_);

  // 2) projections: x @ W.T  (waves = (M/64)*(N/64), 8 waves/block)
  gemm_bf16_wmma<bf16_t><<<(64 * 32) / 8, 256, 0, stream>>>(qbf, Wqbf, qpbf, M, 2048, D_);
  gemm_bf16_wmma<bf16_t><<<(64 * 16) / 8, 256, 0, stream>>>(kbf, Wkbf, kpbf, M, 1024, D_);
  gemm_bf16_wmma<bf16_t><<<(64 *  8) / 8, 256, 0, stream>>>(vbf, Wvbf, vpbf, M,  512, D_);

  // 3) fused differential flash-attention: B*H*(T/16) = 4096 waves
  diff_attn_kernel<<<(B_ * H_ * (T_ / 16)) / 8, 256, 0, stream>>>(qpbf, kpbf, vpbf, lam, attnf);

  // 4) RMSNorm + gain + (1-lambda_init), to bf16
  rmsnorm_scale_kernel<<<M / 8, 256, 0, stream>>>(attnf, g, attnbf);

  // 5) output projection -> f32 d_out
  gemm_bf16_wmma<float><<<(64 * 16) / 8, 256, 0, stream>>>(attnbf, Wobf, out, M, 1024, D_);
}
